// VectorQuantizer_23098334118239
// MI455X (gfx1250) — compile-verified, matrix operand order checked by reading, not run on hardware
//
#include <hip/hip_runtime.h>
#include <hip/hip_bf16.h>

// VQ-VAE vector quantizer for MI455X (gfx1250, wave32, WMMA).
// N=65536 rows, D=64, K=1024 codes.
// Distance GEMM via v_wmma_f32_16x16x32_f16 (f16 in, f32 accumulate) for
// argmin selection; quantized/loss/indices from exact f32 codebook rows.
// Each wave owns TWO 16-row blocks so every LDS B-tile feeds 4 WMMAs.

typedef __attribute__((ext_vector_type(16))) _Float16 v16h;
typedef __attribute__((ext_vector_type(4)))  _Float16 v4h;
typedef __attribute__((ext_vector_type(8)))  float    v8f;

#define VQ_N 65536
#define VQ_D 64
#define VQ_K 1024
#define VQ_QN (VQ_N * VQ_D)          // 4194304
#define VQ_TPB 128                    // 4 waves per block
#define VQ_WAVES_PER_BLOCK 4
#define VQ_ROWS_PER_WAVE 32           // two 16-row WMMA blocks
#define VQ_BLOCKS 512                 // 512 * 4 * 32 = 65536 rows
#define VQ_NPART (VQ_BLOCKS * VQ_WAVES_PER_BLOCK)  // 2048 partial sums
// dynamic LDS: f16 codebook (131072 B) + transposed half-sq-norms (4096 B)
#define VQ_SMEM_BYTES (VQ_K * VQ_D * 2 + VQ_K * 4)

__global__ __launch_bounds__(VQ_TPB, 1) void vq_main_kernel(
    const float* __restrict__ X,   // [N, 64] inputs
    const float* __restrict__ E,   // [K, 64] embeddings (f32)
    float* __restrict__ Q,         // [N, 64] quantized out
    float* __restrict__ IDXO,      // [N] indices out (as float)
    float* __restrict__ partial)   // [VQ_NPART] per-wave sq-err sums
{
    extern __shared__ __align__(16) char smem[];
    _Float16* sEH  = (_Float16*)smem;                  // [K*D] f16 codebook
    float*    sH2T = (float*)(smem + VQ_K * VQ_D * 2); // [16][64] 0.5*||e||^2,
                                                       // transposed: [r][t]

    const int tid  = threadIdx.x;
    const int wave = tid >> 5;
    const int lane = tid & 31;
    const int r    = lane & 15;   // row/col-within-tile index
    const int g    = lane >> 4;   // lane group (K-slice selector)

    // ---- Stage codebook into LDS: f32 -> f16, vectorized (4 halfs / 8B store)
    for (int p = tid * 4; p < VQ_K * VQ_D; p += VQ_TPB * 4) {
        const float4 v = *(const float4*)(E + p);
        v4h hv = { (_Float16)v.x, (_Float16)v.y, (_Float16)v.z, (_Float16)v.w };
        *(v4h*)(sEH + p) = hv;
    }
    // ---- 0.5 * squared norms, stored transposed so each lane's h2 for 4
    //      consecutive column tiles is one contiguous float4 in LDS.
    for (int k = tid; k < VQ_K; k += VQ_TPB) {
        const float4* ev = (const float4*)(E + k * VQ_D);
        float s = 0.0f;
        #pragma unroll
        for (int i = 0; i < 16; ++i) {
            float4 v = ev[i];
            s += v.x * v.x + v.y * v.y + v.z * v.z + v.w * v.w;
        }
        sH2T[(k & 15) * 64 + (k >> 4)] = 0.5f * s;   // [r][t]
    }
    __syncthreads();

    // ---- Load this wave's 32 input rows as two WMMA A operands (f32->f16).
    // Lane (g, r) holds x[row][c*32 + g*16 .. +15] for K-chunk c. A and B
    // share the same per-lane K ordering, so the dot is exact w.r.t. layout
    // (K-permutation invariance of the reduction).
    const int rowBase = (blockIdx.x * VQ_WAVES_PER_BLOCK + wave) * VQ_ROWS_PER_WAVE;
    v16h a[2][2];   // [row-block][K-chunk]
    #pragma unroll
    for (int rb = 0; rb < 2; ++rb) {
        const float* xrow = X + (size_t)(rowBase + rb * 16 + r) * VQ_D;
        #pragma unroll
        for (int c = 0; c < 2; ++c) {
            const float4* xv = (const float4*)(xrow + c * 32 + g * 16);
            #pragma unroll
            for (int q4 = 0; q4 < 4; ++q4) {
                float4 v = xv[q4];
                a[rb][c][q4 * 4 + 0] = (_Float16)v.x;
                a[rb][c][q4 * 4 + 1] = (_Float16)v.y;
                a[rb][c][q4 * 4 + 2] = (_Float16)v.z;
                a[rb][c][q4 * 4 + 3] = (_Float16)v.w;
            }
        }
    }

    // ---- Sweep 64 column tiles, 4 per iteration; each loaded B tile feeds
    // 4 WMMAs (2 row blocks x 2 K-chunks), amortizing DS latency/waits.
    // score[n] = x.e_n - 0.5*||e_n||^2  (maximize <=> minimize L2 distance)
    float best[2][8];
    int   bidx[2][8];
    #pragma unroll
    for (int rb = 0; rb < 2; ++rb)
        #pragma unroll
        for (int j = 0; j < 8; ++j) { best[rb][j] = -3.0e38f; bidx[rb][j] = 0; }

    for (int t4 = 0; t4 < (VQ_K / 16) / 4; ++t4) {
        const int tbase = t4 * 4;
        const float4 h4 = *(const float4*)(sH2T + r * 64 + tbase);

        v16h b0[4], b1[4];
        #pragma unroll
        for (int u = 0; u < 4; ++u) {
            const _Float16* ep = sEH + ((tbase + u) * 16 + r) * VQ_D;
            b0[u] = *(const v16h*)(ep + g * 16);        // K 0..31 slice
            b1[u] = *(const v16h*)(ep + 32 + g * 16);   // K 32..63 slice
        }

        v8f acc[2][4];
        #pragma unroll
        for (int u = 0; u < 4; ++u) {
            #pragma unroll
            for (int rb = 0; rb < 2; ++rb) {
                v8f c8 = {};
                c8 = __builtin_amdgcn_wmma_f32_16x16x32_f16(
                    false, a[rb][0], false, b0[u], (short)0, c8, false, false);
                acc[rb][u] = __builtin_amdgcn_wmma_f32_16x16x32_f16(
                    false, a[rb][1], false, b1[u], (short)0, c8, false, false);
            }
        }

        #pragma unroll
        for (int u = 0; u < 4; ++u) {
            const float h2 = (u == 0) ? h4.x : (u == 1) ? h4.y
                           : (u == 2) ? h4.z : h4.w;
            const int col = (tbase + u) * 16 + r;
            #pragma unroll
            for (int rb = 0; rb < 2; ++rb) {
                #pragma unroll
                for (int j = 0; j < 8; ++j) {
                    float s = acc[rb][u][j] - h2;  // row j+8*g of block rb
                    if (s > best[rb][j]) { best[rb][j] = s; bidx[rb][j] = col; }
                }
            }
        }
    }

    // ---- Reduce winners across the 16 lanes holding each row (xor 8,4,2,1
    // stays inside each 16-lane group on wave32). Ties -> smaller index.
    #pragma unroll
    for (int m = 8; m >= 1; m >>= 1) {
        #pragma unroll
        for (int rb = 0; rb < 2; ++rb) {
            #pragma unroll
            for (int j = 0; j < 8; ++j) {
                float os = __shfl_xor(best[rb][j], m, 32);
                int   oi = __shfl_xor(bidx[rb][j], m, 32);
                if (os > best[rb][j] ||
                    (os == best[rb][j] && oi < bidx[rb][j])) {
                    best[rb][j] = os; bidx[rb][j] = oi;
                }
            }
        }
    }

    // ---- Emit: gather exact f32 code rows, write quantized + index, sum err.
    float acc2 = 0.0f;
    #pragma unroll
    for (int rb = 0; rb < 2; ++rb) {
        #pragma unroll
        for (int j = 0; j < 8; ++j) {
            #pragma unroll
            for (int g2 = 0; g2 < 2; ++g2) {
                const int idx = __shfl(bidx[rb][j], g2 * 16, 32); // uniform/row
                const int row = rowBase + rb * 16 + j + 8 * g2;
                const float2 qv = ((const float2*)(E + (size_t)idx * VQ_D))[lane];
                const float2 xv = ((const float2*)(X + (size_t)row * VQ_D))[lane];
                float d0 = qv.x - xv.x, d1 = qv.y - xv.y;
                acc2 += d0 * d0 + d1 * d1;
                ((float2*)(Q + (size_t)row * VQ_D))[lane] = qv;
                if (lane == 0) IDXO[row] = (float)idx;
            }
        }
    }
    #pragma unroll
    for (int m = 16; m >= 1; m >>= 1) acc2 += __shfl_xor(acc2, m, 32);
    if (lane == 0) partial[blockIdx.x * VQ_WAVES_PER_BLOCK + wave] = acc2;
}

__global__ __launch_bounds__(256) void vq_loss_kernel(
    const float* __restrict__ partial, float* __restrict__ out_loss)
{
    __shared__ float red[256];
    float s = 0.0f;
    // fixed order per thread + fixed tree -> deterministic across replays
    for (int i = threadIdx.x; i < VQ_NPART; i += 256) s += partial[i];
    red[threadIdx.x] = s;
    __syncthreads();
    for (int w = 128; w > 0; w >>= 1) {
        if (threadIdx.x < w) red[threadIdx.x] += red[threadIdx.x + w];
        __syncthreads();
    }
    if (threadIdx.x == 0) {
        // q_latent_loss == e_latent_loss in value -> loss = 1.25 * MSE
        out_loss[0] = 1.25f * red[0] / (float)VQ_QN;
    }
}

extern "C" void kernel_launch(void* const* d_in, const int* in_sizes, int n_in,
                              void* d_out, int out_size, void* d_ws, size_t ws_size,
                              hipStream_t stream) {
    (void)in_sizes; (void)n_in; (void)out_size; (void)ws_size;
    const float* X = (const float*)d_in[0];   // [16,64,64,64] f32
    const float* E = (const float*)d_in[1];   // [1024,64] f32
    float* out  = (float*)d_out;
    float* Q    = out;                 // quantized: [0, 4194304)
    float* loss = out + VQ_QN;         // scalar at 4194304
    float* IDXO = out + VQ_QN + 1;     // indices (as float): [4194305, +65536)
    float* partial = (float*)d_ws;     // 2048 floats of scratch

    vq_main_kernel<<<VQ_BLOCKS, VQ_TPB, VQ_SMEM_BYTES, stream>>>(X, E, Q, IDXO, partial);
    vq_loss_kernel<<<1, 256, 0, stream>>>(partial, loss);
}
